// BaselineDenseMamba_11158325035373
// MI455X (gfx1250) — matrix operand order repaired and checked by reading, not compiled, not run on hardware
//
#include <hip/hip_runtime.h>
#include <hip/hip_bf16.h>

// ---------------------------------------------------------------------------
// Mamba block reduced to: GEMM1(bf16 WMMA) + fused channel-gate/SiLU epilogue
// -> bf16 intermediate -> GEMM2(bf16 WMMA) + bias.  Memory-bound (~1 GB HBM
// traffic => ~45us floor at 23.3 TB/s).  bf16 tiles are streamed into LDS
// with gfx1250 GLOBAL_LOAD_ASYNC_TO_LDS_B128 (ASYNCcnt) and double-buffered;
// GEMM1's A stream stays on the VGPR path because it needs f32->bf16 cvt.
// ---------------------------------------------------------------------------

typedef __attribute__((ext_vector_type(16))) __bf16 v16bf;
typedef __attribute__((ext_vector_type(8)))  __bf16 v8bf;
typedef __attribute__((ext_vector_type(8)))  float  v8f;

#define D_MODEL 2048
#define D_INNER 4096
#define BATCH   32768

#define BM 128
#define BN 128
#define BK 32
#define LDT 40   // padded LDS row stride (bf16 elems): 80B rows, 16B aligned

union FragU { v16bf v; v8bf h[2]; };

__device__ __forceinline__ v8bf cvt8(float4 a, float4 b) {
  v8bf r;
  r[0] = (__bf16)a.x; r[1] = (__bf16)a.y; r[2] = (__bf16)a.z; r[3] = (__bf16)a.w;
  r[4] = (__bf16)b.x; r[5] = (__bf16)b.y; r[6] = (__bf16)b.z; r[7] = (__bf16)b.w;
  return r;
}

// gfx1250 async global->LDS copy, 16B per lane; tracked by ASYNCcnt.
// ldsoff = low 32 bits of flat shared address (LDS byte address per ISA 10.2).
__device__ __forceinline__ void async_ld_b128(const void* gptr, unsigned ldsoff) {
  asm volatile("global_load_async_to_lds_b128 %0, %1, off"
               :: "v"(ldsoff), "v"(gptr)
               : "memory");
}
__device__ __forceinline__ void wait_async0() {
  asm volatile("s_wait_asynccnt 0x0" ::: "memory");
}
__device__ __forceinline__ unsigned lds_off(const void* p) {
  return (unsigned)(unsigned long long)p;   // flat->LDS: addr[31:0]
}

// ---- per-channel coefficient precompute -----------------------------------
// s = softplus(dt)*sum_n(B*C) + D + conv_w[:, -1];  cb = conv_b; z = in_b
__global__ void mamba_coef(const float* __restrict__ Bm, const float* __restrict__ Cm,
                           const float* __restrict__ Dp, const float* __restrict__ dt,
                           const float* __restrict__ conv_w, const float* __restrict__ conv_b,
                           const float* __restrict__ in_b, float4* __restrict__ coef) {
  int d = blockIdx.x * blockDim.x + threadIdx.x;
  if (d >= D_INNER) return;
  float bc = 0.f;
#pragma unroll
  for (int n = 0; n < 16; ++n) bc += Bm[d * 16 + n] * Cm[d * 16 + n];
  float t = dt[d];
  float delta = (t > 20.f) ? t : log1pf(__expf(t));
  float g = delta * bc + Dp[d];
  float s = g + conv_w[d * 4 + 3];           // only last conv tap contributes
  coef[d] = make_float4(s, conv_b[d], in_b[d], 0.f);
}

// ---- one-time weight transpose + f32->bf16 downconvert --------------------
// in: (R x C) f32 row-major  ->  out: (C x R) bf16 row-major
__global__ void transpose_f32_to_bf16(const float* __restrict__ in,
                                      __bf16* __restrict__ out, int R, int C) {
  __shared__ float tile[32][33];
  int c = blockIdx.x * 32 + threadIdx.x;
  int r0 = blockIdx.y * 32;
#pragma unroll
  for (int j = 0; j < 4; ++j)
    tile[threadIdx.y + 8 * j][threadIdx.x] = in[(size_t)(r0 + threadIdx.y + 8 * j) * C + c];
  __syncthreads();
  int c0 = blockIdx.x * 32;
#pragma unroll
  for (int j = 0; j < 4; ++j)
    out[(size_t)(c0 + threadIdx.y + 8 * j) * R + r0 + threadIdx.x] =
        (__bf16)tile[threadIdx.x][threadIdx.y + 8 * j];
}

// ---- GEMM1: gated = silu((x @ in_w + in_b) * s + cb), bf16 out ------------
__global__ __launch_bounds__(256) void mamba_gemm1(
    const float* __restrict__ X,        // 32768 x 2048 f32
    const __bf16* __restrict__ W1T,     // 4096 x 2048 bf16 (N-major)
    const float4* __restrict__ coef,    // 4096
    __bf16* __restrict__ gated) {       // 32768 x 4096 bf16
  __shared__ __bf16 Alds[2][BM * LDT];
  __shared__ __bf16 Blds[2][BN * LDT];
  const int m0 = blockIdx.y * BM, n0 = blockIdx.x * BN;
  const int tid = threadIdx.x;
  const int lrow = tid >> 1, lcol = (tid & 1) * 16;

  const float*  ag = X   + (size_t)(m0 + lrow) * D_MODEL + lcol;
  const __bf16* bg = W1T + (size_t)(n0 + lrow) * D_MODEL + lcol;
  __bf16* as0 = &Alds[0][lrow * LDT + lcol];
  __bf16* as1 = &Alds[1][lrow * LDT + lcol];
  const unsigned bo0 = lds_off(&Blds[0][lrow * LDT + lcol]);
  const unsigned bo1 = lds_off(&Blds[1][lrow * LDT + lcol]);

  const int lane = tid & 31;
  const int wm = ((tid >> 5) >> 2) * 64;   // wave row offset (2 wave-rows)
  const int wn = ((tid >> 5) & 3) * 32;    // wave col offset (4 wave-cols)
  const int lr = lane & 15;
  const int ka  = (lane >> 4) * 8;         // A frag K offset per ISA layout
  const int kb2 = (lane >> 4) * 8;         // B frag K offset per ISA layout

  v8f zero = {0.f, 0.f, 0.f, 0.f, 0.f, 0.f, 0.f, 0.f};
  v8f acc[4][2];
#pragma unroll
  for (int i = 0; i < 4; ++i)
#pragma unroll
    for (int j = 0; j < 2; ++j) acc[i][j] = zero;

  {  // prologue: fill buffer 0 (A: load+cvt+store; B: async DMA)
    async_ld_b128(bg,     bo0);
    async_ld_b128(bg + 8, bo0 + 16);
    const float4* p = (const float4*)ag;
    float4 f0 = p[0], f1 = p[1], f2 = p[2], f3 = p[3];
    *(v8bf*)(as0)     = cvt8(f0, f1);
    *(v8bf*)(as0 + 8) = cvt8(f2, f3);
  }
  wait_async0();
  __syncthreads();

  const int KB = D_MODEL / BK;  // 64
  for (int kb = 0; kb < KB; ++kb) {
    const int cur = kb & 1;
    const bool pf = (kb + 1 < KB);
    float4 f0, f1, f2, f3;
    if (pf) {  // prefetch next k-tile: B via async DMA, A via VGPR (needs cvt)
      const int kk = (kb + 1) * BK;
      const unsigned bon = cur ? bo0 : bo1;
      async_ld_b128(bg + kk,     bon);
      async_ld_b128(bg + kk + 8, bon + 16);
      const float4* p = (const float4*)(ag + kk);
      f0 = p[0]; f1 = p[1]; f2 = p[2]; f3 = p[3];
    }
    FragU af[4], bfr[2];
#pragma unroll
    for (int mi = 0; mi < 4; ++mi) {
      const __bf16* p = &Alds[cur][(wm + mi * 16 + lr) * LDT + ka];
      af[mi].h[0] = *(const v8bf*)(p);
      af[mi].h[1] = *(const v8bf*)(p + 16);
    }
#pragma unroll
    for (int ni = 0; ni < 2; ++ni) {
      const __bf16* p = &Blds[cur][(wn + ni * 16 + lr) * LDT + kb2];
      bfr[ni].h[0] = *(const v8bf*)(p);
      bfr[ni].h[1] = *(const v8bf*)(p + 16);
    }
#pragma unroll
    for (int mi = 0; mi < 4; ++mi)
#pragma unroll
      for (int ni = 0; ni < 2; ++ni)
        acc[mi][ni] = __builtin_amdgcn_wmma_f32_16x16x32_bf16(
            false, af[mi].v, false, bfr[ni].v, (short)0, acc[mi][ni], false, false);
    if (pf) {
      __bf16* asn = cur ? as0 : as1;
      *(v8bf*)(asn)     = cvt8(f0, f1);
      *(v8bf*)(asn + 8) = cvt8(f2, f3);
    }
    wait_async0();
    __syncthreads();
  }

  // fused epilogue: xi = acc + in_b; h = xi*s + cb; silu(h); store bf16
#pragma unroll
  for (int mi = 0; mi < 4; ++mi)
#pragma unroll
    for (int ni = 0; ni < 2; ++ni) {
      const int ng = n0 + wn + ni * 16 + lr;
      const float4 cf = coef[ng];
      const int mg = m0 + wm + mi * 16 + ka;  // ka==8 for lanes 16..31 (D rows M=8..15)
      __bf16* op = gated + (size_t)mg * D_INNER + ng;
#pragma unroll
      for (int r = 0; r < 8; ++r) {
        float xi = acc[mi][ni][r] + cf.z;
        float h  = fmaf(xi, cf.x, cf.y);
        float sg = h * __builtin_amdgcn_rcpf(1.0f + __expf(-h));
        op[(size_t)r * D_INNER] = (__bf16)sg;
      }
    }
}

// ---- GEMM2: out = gated @ out_w + out_b, f32 out --------------------------
// Both operand streams are bf16 -> fully async global->LDS DMA, no staging.
__global__ __launch_bounds__(256) void mamba_gemm2(
    const __bf16* __restrict__ G,       // 32768 x 4096 bf16
    const __bf16* __restrict__ W2T,     // 2048 x 4096 bf16 (N-major)
    const float* __restrict__ out_b,    // 2048
    float* __restrict__ out) {          // 32768 x 2048 f32
  __shared__ __bf16 Alds[2][BM * LDT];
  __shared__ __bf16 Blds[2][BN * LDT];
  const int m0 = blockIdx.y * BM, n0 = blockIdx.x * BN;
  const int tid = threadIdx.x;
  const int lrow = tid >> 1, lcol = (tid & 1) * 16;

  const __bf16* ag = G   + (size_t)(m0 + lrow) * D_INNER + lcol;
  const __bf16* bg = W2T + (size_t)(n0 + lrow) * D_INNER + lcol;
  const unsigned ao0 = lds_off(&Alds[0][lrow * LDT + lcol]);
  const unsigned ao1 = lds_off(&Alds[1][lrow * LDT + lcol]);
  const unsigned bo0 = lds_off(&Blds[0][lrow * LDT + lcol]);
  const unsigned bo1 = lds_off(&Blds[1][lrow * LDT + lcol]);

  const int lane = tid & 31;
  const int wm = ((tid >> 5) >> 2) * 64;
  const int wn = ((tid >> 5) & 3) * 32;
  const int lr = lane & 15;
  const int ka  = (lane >> 4) * 8;
  const int kb2 = (lane >> 4) * 8;

  v8f zero = {0.f, 0.f, 0.f, 0.f, 0.f, 0.f, 0.f, 0.f};
  v8f acc[4][2];
#pragma unroll
  for (int i = 0; i < 4; ++i)
#pragma unroll
    for (int j = 0; j < 2; ++j) acc[i][j] = zero;

  {  // prologue: fill buffer 0 entirely via async DMA
    async_ld_b128(ag,     ao0);
    async_ld_b128(ag + 8, ao0 + 16);
    async_ld_b128(bg,     bo0);
    async_ld_b128(bg + 8, bo0 + 16);
  }
  wait_async0();
  __syncthreads();

  const int KB = D_INNER / BK;  // 128
  for (int kb = 0; kb < KB; ++kb) {
    const int cur = kb & 1;
    if (kb + 1 < KB) {
      const int kk = (kb + 1) * BK;
      const unsigned aon = cur ? ao0 : ao1;
      const unsigned bon = cur ? bo0 : bo1;
      async_ld_b128(ag + kk,     aon);
      async_ld_b128(ag + kk + 8, aon + 16);
      async_ld_b128(bg + kk,     bon);
      async_ld_b128(bg + kk + 8, bon + 16);
    }
    FragU af[4], bfr[2];
#pragma unroll
    for (int mi = 0; mi < 4; ++mi) {
      const __bf16* p = &Alds[cur][(wm + mi * 16 + lr) * LDT + ka];
      af[mi].h[0] = *(const v8bf*)(p);
      af[mi].h[1] = *(const v8bf*)(p + 16);
    }
#pragma unroll
    for (int ni = 0; ni < 2; ++ni) {
      const __bf16* p = &Blds[cur][(wn + ni * 16 + lr) * LDT + kb2];
      bfr[ni].h[0] = *(const v8bf*)(p);
      bfr[ni].h[1] = *(const v8bf*)(p + 16);
    }
#pragma unroll
    for (int mi = 0; mi < 4; ++mi)
#pragma unroll
      for (int ni = 0; ni < 2; ++ni)
        acc[mi][ni] = __builtin_amdgcn_wmma_f32_16x16x32_bf16(
            false, af[mi].v, false, bfr[ni].v, (short)0, acc[mi][ni], false, false);
    wait_async0();
    __syncthreads();
  }

#pragma unroll
  for (int mi = 0; mi < 4; ++mi)
#pragma unroll
    for (int ni = 0; ni < 2; ++ni) {
      const int ng = n0 + wn + ni * 16 + lr;
      const float bias = out_b[ng];
      const int mg = m0 + wm + mi * 16 + ka;
      float* op = out + (size_t)mg * D_MODEL + ng;
#pragma unroll
      for (int r = 0; r < 8; ++r)
        op[(size_t)r * D_MODEL] = acc[mi][ni][r] + bias;
    }
}

// ---------------------------------------------------------------------------
extern "C" void kernel_launch(void* const* d_in, const int* in_sizes, int n_in,
                              void* d_out, int out_size, void* d_ws, size_t ws_size,
                              hipStream_t stream) {
  const float* x      = (const float*)d_in[0];
  const float* in_w   = (const float*)d_in[1];   // 2048 x 4096
  const float* in_b   = (const float*)d_in[2];
  const float* conv_w = (const float*)d_in[3];   // 4096 x 4
  const float* conv_b = (const float*)d_in[4];
  // d_in[5] = A_log: unused (exp(delta*A)*h0 term vanishes with h0 = 0)
  const float* Bm     = (const float*)d_in[6];
  const float* Cm     = (const float*)d_in[7];
  const float* Dp     = (const float*)d_in[8];
  const float* dt     = (const float*)d_in[9];
  const float* out_w  = (const float*)d_in[10];  // 4096 x 2048
  const float* out_b  = (const float*)d_in[11];

  // workspace layout: coef 64KB | W1T bf16 16MB | W2T bf16 16MB | gated bf16 256MB
  char* ws = (char*)d_ws;
  float4* coef  = (float4*)ws;
  __bf16* w1t   = (__bf16*)(ws + (64ull << 10));
  __bf16* w2t   = (__bf16*)(ws + (64ull << 10) + (16ull << 20));
  __bf16* gated = (__bf16*)(ws + (64ull << 10) + (32ull << 20));

  mamba_coef<<<D_INNER / 256, 256, 0, stream>>>(Bm, Cm, Dp, dt, conv_w, conv_b, in_b, coef);
  transpose_f32_to_bf16<<<dim3(D_INNER / 32, D_MODEL / 32), dim3(32, 8), 0, stream>>>(
      in_w, w1t, D_MODEL, D_INNER);
  transpose_f32_to_bf16<<<dim3(D_MODEL / 32, D_INNER / 32), dim3(32, 8), 0, stream>>>(
      out_w, w2t, D_INNER, D_MODEL);
  mamba_gemm1<<<dim3(D_INNER / BN, BATCH / BM), 256, 0, stream>>>(x, w1t, coef, gated);
  mamba_gemm2<<<dim3(D_MODEL / BN, BATCH / BM), 256, 0, stream>>>(gated, w2t, out_b,
                                                                  (float*)d_out);
}
